// Propagation_16673063043307
// MI455X (gfx1250) — matrix operand order repaired and checked by reading, not compiled, not run on hardware
//
#include <hip/hip_runtime.h>
#include <hip/hip_bf16.h>

// ---------------------------------------------------------------------------
// Fused graph-propagation for MI455X (gfx1250, wave32, WMMA + async LDS DMA).
//   scores = val @ val^T            (bf16 WMMA, f32 acc)
//   edges  = softsign(scores)       (VALU, never touches memory)
//   pv_ext = [val @ Wv | state | 0] (precomputed bf16, stored transposed)
//   out    = edges @ pv_ext         (bf16 WMMA; col 256 = delta_state)
// Shared key tiles are staged global->LDS with double-buffered
// global_load_async_to_lds_b128 (ASYNCcnt), compute reads via ds_load_b128.
// ---------------------------------------------------------------------------

typedef __bf16 bf16_t;
typedef __attribute__((ext_vector_type(16))) __bf16 v16bf;
typedef __attribute__((ext_vector_type(8)))  __bf16 v8bf;
typedef __attribute__((ext_vector_type(4)))  __bf16 v4bf;
typedef __attribute__((ext_vector_type(8)))  float  v8f;
typedef __attribute__((ext_vector_type(4)))  float  v4f;
typedef __attribute__((ext_vector_type(4)))  int    v4i;

#define B_ 4
#define N_ 4096
#define D_ 256
#define E_ 288   // D + 32 pad; row 256 of pvT carries `state`, rest zero
#define NT_ (N_ / 32)

typedef __attribute__((address_space(1))) v4i as1_v4i;
typedef __attribute__((address_space(3))) v4i as3_v4i;

union BF16x16 { v16bf v; v8bf h[2]; bf16_t e[16]; };

__device__ __forceinline__ v8f zero8() {
    v8f z = {0.f, 0.f, 0.f, 0.f, 0.f, 0.f, 0.f, 0.f};
    return z;
}

__device__ __forceinline__ void async_b128(const void* g, void* l) {
    __builtin_amdgcn_global_load_async_to_lds_b128(
        (as1_v4i*)(v4i*)g, (as3_v4i*)(v4i*)l, 0, 0);
}

// --------------------------- elementwise prep ------------------------------

__global__ void k_cast(const float* __restrict__ val, bf16_t* __restrict__ valb) {
    int i = blockIdx.x * blockDim.x + threadIdx.x;   // B*N*D/4 threads
    v4f f = ((const v4f*)val)[i];
    v4bf o;
    #pragma unroll
    for (int j = 0; j < 4; ++j) o[j] = (bf16_t)f[j];
    ((v4bf*)valb)[i] = o;
}

__global__ void k_fill(const float* __restrict__ state, bf16_t* __restrict__ pvT) {
    int i = blockIdx.x * blockDim.x + threadIdx.x;   // B*32*N threads
    int m   = i & (N_ - 1);
    int row = (i >> 12) & 31;                        // padded row 0..31
    int b   = i >> 17;
    bf16_t v = (row == 0) ? (bf16_t)state[b * N_ + m] : (bf16_t)0.f;
    pvT[((size_t)b * E_ + 256 + row) * N_ + m] = v;
}

// --------------------- projected_val = val @ Wv (WMMA) ---------------------

__global__ __launch_bounds__(128)
void k_proj(const bf16_t* __restrict__ valb, const float* __restrict__ Wv,
            bf16_t* __restrict__ pvT) {
    const int lane = threadIdx.x & 31;
    const int wave = threadIdx.x >> 5;
    const int r    = lane & 15;
    const int half = lane >> 4;

    int tile = blockIdx.x * 4 + wave;        // B * (N/16) * (D/16) tiles
    int et = tile & 15;
    int nt = (tile >> 4) & 255;
    int b  = tile >> 12;
    int n0 = nt * 16, e0 = et * 16;

    const bf16_t* valB = valb + (size_t)b * N_ * D_;
    v8f acc = zero8();

    for (int d0 = 0; d0 < D_; d0 += 32) {
        BF16x16 a;   // val rows n0+r, K = d0 + half*16 .. +15
        const bf16_t* ap = valB + (size_t)(n0 + r) * D_ + d0 + half * 16;
        a.h[0] = *(const v8bf*)(ap);
        a.h[1] = *(const v8bf*)(ap + 8);
        BF16x16 bb;  // lane holds column e0+r, K = d0 + half*16 .. +15
        #pragma unroll
        for (int j = 0; j < 16; ++j)
            bb.e[j] = (bf16_t)Wv[(size_t)(d0 + half * 16 + j) * D_ + e0 + r];
        acc = __builtin_amdgcn_wmma_f32_16x16x32_bf16(false, a.v, false, bb.v,
                                                      (short)0, acc, false, false);
    }
    #pragma unroll
    for (int v = 0; v < 8; ++v) {            // transposed store pvT[b, e, m]
        int row = v + half * 8;
        pvT[((size_t)b * E_ + e0 + r) * N_ + n0 + row] = (bf16_t)acc[v];
    }
}

// ------------------------------ fused main ---------------------------------

__device__ __forceinline__ void stage_tiles(const bf16_t* valB, const bf16_t* pvB,
                                            int m0, bf16_t (*lv)[D_],
                                            bf16_t (*lp)[32], int tid) {
    // val key tile: 32 rows x 256 bf16 = 1024 chunks of 16B, 8 per thread
    #pragma unroll
    for (int k = 0; k < 8; ++k) {
        int c = k * 128 + tid;
        int row = c >> 5, col = (c & 31) * 8;
        async_b128(valB + (size_t)(m0 + row) * D_ + col, &lv[row][col]);
    }
    // pv tile: 288 rows x 32 bf16 = 1152 chunks of 16B, 9 per thread (uniform)
    #pragma unroll
    for (int k = 0; k < 9; ++k) {
        int c = k * 128 + tid;
        int row = c >> 2, col = (c & 3) * 8;
        async_b128(pvB + (size_t)row * N_ + m0 + col, &lp[row][col]);
    }
}

__global__ __launch_bounds__(128)
void k_fused(const bf16_t* __restrict__ valb,   // [B,N,D] bf16
             const bf16_t* __restrict__ pvT,    // [B,E,N] bf16 (transposed)
             float* __restrict__ dstate,        // [B,N]
             float* __restrict__ dval) {        // [B,N,D]
    __shared__ __align__(16) bf16_t lds_val[2][32][D_];   // 2 x 16 KB
    __shared__ __align__(16) bf16_t lds_pv[2][E_][32];    // 2 x 18 KB
    __shared__ __align__(16) bf16_t lds_edges[4][16][32]; // 4 KB

    const int tid  = threadIdx.x;
    const int lane = tid & 31;
    const int wave = tid >> 5;
    const int b    = blockIdx.y;
    const int n0   = blockIdx.x * 64 + wave * 16;
    const int r    = lane & 15;
    const int half = lane >> 4;

    const bf16_t* valB = valb + (size_t)b * N_ * D_;
    const bf16_t* pvB  = pvT  + (size_t)b * E_ * N_;

    // loop-invariant A fragments: val rows n0..n0+15, all 8 K-steps
    BF16x16 afrag[8];
    #pragma unroll
    for (int kd = 0; kd < 8; ++kd) {
        const bf16_t* ap = valB + (size_t)(n0 + r) * D_ + kd * 32 + half * 16;
        afrag[kd].h[0] = *(const v8bf*)(ap);
        afrag[kd].h[1] = *(const v8bf*)(ap + 8);
    }

    v8f acc[17];
    #pragma unroll
    for (int t = 0; t < 17; ++t) acc[t] = zero8();

    stage_tiles(valB, pvB, 0, lds_val[0], lds_pv[0], tid);

    for (int it = 0; it < NT_; ++it) {
        bf16_t (*lv)[D_] = lds_val[it & 1];
        bf16_t (*lp)[32] = lds_pv[it & 1];
        if (it + 1 < NT_) {   // overlap next tile's DMA with this tile's math
            stage_tiles(valB, pvB, (it + 1) * 32,
                        lds_val[(it + 1) & 1], lds_pv[(it + 1) & 1], tid);
            __builtin_amdgcn_s_wait_asynccnt(17);  // prev tile (17 ops) done
        } else {
            __builtin_amdgcn_s_wait_asynccnt(0);
        }
        __syncthreads();

        // ---- GEMM1: S[16x32] = val[n0:16,:] @ key_tile^T, K = 256
        v8f s0 = zero8(), s1 = zero8();
        #pragma unroll
        for (int kd = 0; kd < 8; ++kd) {
            int d0 = kd * 32;
            BF16x16 b0, b1;   // columns of val^T = LDS rows (contiguous)
            const bf16_t* bp0 = &lv[r][d0 + half * 16];
            b0.h[0] = *(const v8bf*)(bp0);
            b0.h[1] = *(const v8bf*)(bp0 + 8);
            const bf16_t* bp1 = &lv[16 + r][d0 + half * 16];
            b1.h[0] = *(const v8bf*)(bp1);
            b1.h[1] = *(const v8bf*)(bp1 + 8);
            s0 = __builtin_amdgcn_wmma_f32_16x16x32_bf16(false, afrag[kd].v, false,
                                                         b0.v, (short)0, s0, false, false);
            s1 = __builtin_amdgcn_wmma_f32_16x16x32_bf16(false, afrag[kd].v, false,
                                                         b1.v, (short)0, s1, false, false);
        }

        // ---- softsign; bounce C-layout -> A-layout through per-wave LDS
        #pragma unroll
        for (int v = 0; v < 8; ++v) {
            int row = v + half * 8;
            float e0 = s0[v]; e0 = e0 / (1.0f + __builtin_fabsf(e0));
            float e1 = s1[v]; e1 = e1 / (1.0f + __builtin_fabsf(e1));
            lds_edges[wave][row][r]      = (bf16_t)e0;
            lds_edges[wave][row][16 + r] = (bf16_t)e1;
        }
        __builtin_amdgcn_wave_barrier();
        BF16x16 ea;
        ea.h[0] = *(const v8bf*)(&lds_edges[wave][r][half * 16]);
        ea.h[1] = *(const v8bf*)(&lds_edges[wave][r][half * 16 + 8]);
        __builtin_amdgcn_wave_barrier();

        // ---- GEMM2: acc[t] += edges[16x32] @ pv_ext[m:32, t*16:+16]
        #pragma unroll
        for (int t = 0; t < 17; ++t) {
            BF16x16 bb;
            const bf16_t* pp = &lp[t * 16 + r][half * 16];
            bb.h[0] = *(const v8bf*)(pp);
            bb.h[1] = *(const v8bf*)(pp + 8);
            acc[t] = __builtin_amdgcn_wmma_f32_16x16x32_bf16(false, ea.v, false, bb.v,
                                                             (short)0, acc[t], false, false);
        }
        __syncthreads();   // all waves done with this buffer before reuse
    }

    // ---- epilogue: tiles 0..15 -> delta_val, tile 16 col 0 -> delta_state
    #pragma unroll
    for (int t = 0; t < 16; ++t) {
        #pragma unroll
        for (int v = 0; v < 8; ++v) {
            int row = v + half * 8;
            dval[((size_t)b * N_ + n0 + row) * D_ + t * 16 + r] = acc[t][v];
        }
    }
    if (r == 0) {
        #pragma unroll
        for (int v = 0; v < 8; ++v) {
            int row = v + half * 8;
            dstate[(size_t)b * N_ + n0 + row] = acc[16][v];
        }
    }
}

// ------------------------------- launcher ----------------------------------

extern "C" void kernel_launch(void* const* d_in, const int* in_sizes, int n_in,
                              void* d_out, int out_size, void* d_ws, size_t ws_size,
                              hipStream_t stream) {
    const float* val   = (const float*)d_in[0];   // [B,N,D]
    const float* state = (const float*)d_in[1];   // [B,N]
    const float* Wv    = (const float*)d_in[2];   // [D,D]

    float* dstate = (float*)d_out;                       // [B,N]
    float* dval   = dstate + (size_t)B_ * N_;            // [B,N,D]

    bf16_t* valb = (bf16_t*)d_ws;                        // [B,N,D]   8.4 MB
    bf16_t* pvT  = valb + (size_t)B_ * N_ * D_;          // [B,E,N]   9.4 MB

    k_cast<<<(B_ * N_ * D_ / 4) / 256, 256, 0, stream>>>(val, valb);
    k_fill<<<(B_ * 32 * N_) / 256, 256, 0, stream>>>(state, pvT);
    k_proj<<<(B_ * (N_ / 16) * (D_ / 16)) / 4, 128, 0, stream>>>(valb, Wv, pvT);
    k_fused<<<dim3(N_ / 64, B_), 128, 0, stream>>>(valb, pvT, dstate, dval);
}
